// CasualMultiheadSelfAttention_43258910605335
// MI455X (gfx1250) — matrix operand order repaired and checked by reading, not compiled, not run on hardware
//
#include <hip/hip_runtime.h>

typedef __attribute__((ext_vector_type(16))) __bf16 v16bf;
typedef __attribute__((ext_vector_type(8)))  float  v8f;

constexpr int D_MODEL   = 1024;
constexpr int NUM_HEADS = 16;
constexpr int D_K       = 64;
constexpr int Bsz       = 2;
constexpr int Lseq      = 2048;
constexpr int Mrows     = Bsz * Lseq;   // 4096 rows of (b,l)

__device__ __forceinline__ unsigned short f2bf(float f) {
  unsigned u = __builtin_bit_cast(unsigned, f);
  unsigned r = u + 0x7FFFu + ((u >> 16) & 1u);   // round-to-nearest-even
  return (unsigned short)(r >> 16);
}

union FragU {              // 16 bf16 = 32 bytes = two 128-bit loads
  uint4 q[2];
  v16bf v;
  unsigned short h[16];
};

// -------- CDNA5 async copy: global -> LDS, 16B per lane, tracked by ASYNCcnt
__device__ __forceinline__ void async_b128(unsigned ldsaddr, const void* gaddr) {
  asm volatile("global_load_async_to_lds_b128 %0, %1, off"
               :: "v"(ldsaddr), "v"(gaddr) : "memory");
}
__device__ __forceinline__ void wait_async0() {
  asm volatile("s_wait_asynccnt 0" ::: "memory");
}
__device__ __forceinline__ unsigned lds_off(const void* p) {
  return (unsigned)(unsigned long long)p;  // low 32b of generic addr = LDS offset
}

// ---------------------------------------------------------------- cast
__global__ void cast_f32_bf16(const float* __restrict__ src,
                              unsigned short* __restrict__ dst, int n) {
  int i = blockIdx.x * blockDim.x + threadIdx.x;
  if (i < n) dst[i] = f2bf(src[i]);
}

// ---------------------------------------------------------------- GEMM
// C[m,n] = sum_k A[m,k] * W[n,k]; each wave computes a 16x64 tile (A reused 4x).
// mode 0: bf16 row-major | 1: RoPE, bf16 row-major | 2: f32 row-major
// mode 3: bf16 transposed head layout [B,H,D_K,L]  (V projection)
__global__ __launch_bounds__(256)
void gemm_wmma(const unsigned short* __restrict__ A,
               const unsigned short* __restrict__ W,
               void* __restrict__ out,
               const int* __restrict__ pos,
               int mode) {
  const int lane = threadIdx.x & 31;
  const int wave = threadIdx.x >> 5;
  const int col  = lane & 15;        // C-frag: n = col
  const int hi   = lane >> 4;        // C-frag: m = r + hi*8
  const int mtile = blockIdx.x;
  const int ngrp  = blockIdx.y * 8 + wave;   // 0..15, each owns 64 output cols

  const unsigned short* arow = A + (size_t)(mtile * 16 + col) * D_MODEL;
  const unsigned short* wrow[4];
#pragma unroll
  for (int t = 0; t < 4; ++t)
    wrow[t] = W + (size_t)(ngrp * 64 + t * 16 + col) * D_MODEL;

  v8f acc[4] = {{}, {}, {}, {}};
  for (int k = 0; k < D_MODEL; k += 32) {
    FragU a;
    const uint4* pa = (const uint4*)(arow + k + hi * 16);
    a.q[0] = pa[0];
    a.q[1] = pa[1];
#pragma unroll
    for (int t = 0; t < 4; ++t) {
      FragU bfr;
      const uint4* pb = (const uint4*)(wrow[t] + k + hi * 16);
      bfr.q[0] = pb[0];
      bfr.q[1] = pb[1];
      acc[t] = __builtin_amdgcn_wmma_f32_16x16x32_bf16(false, a.v, false, bfr.v,
                                                       (short)0, acc[t], false, false);
    }
  }

#pragma unroll
  for (int t = 0; t < 4; ++t) {
#pragma unroll
    for (int r = 0; r < 8; ++r) {
      const int m = mtile * 16 + r + hi * 8;
      const int n = ngrp * 64 + t * 16 + col;
      float val = acc[t][r];
      float other = __shfl_xor(val, 1);        // RoPE partner (adjacent n lane)
      if (mode == 1) {
        const int d = n & (D_K - 1);
        float freq = __expf(-(float)(d & ~1) * (9.210340371976184f / (float)D_K));
        float ang  = (float)pos[m] * freq;
        float sn, cs;
        __sincosf(ang, &sn, &cs);
        val = ((d & 1) == 0) ? (val * cs - other * sn)   // r_even
                             : (other * sn + val * cs);  // r_odd
      }
      if (mode == 2) {
        ((float*)out)[(size_t)m * D_MODEL + n] = val;
      } else if (mode == 3) {
        const int bb = m >> 11, l = m & (Lseq - 1);      // Lseq = 2048
        const int hh = n >> 6,  d = n & (D_K - 1);
        ((unsigned short*)out)[((size_t)(bb * NUM_HEADS + hh) * D_K + d) * Lseq + l]
            = f2bf(val);
      } else {
        ((unsigned short*)out)[(size_t)m * D_MODEL + n] = f2bf(val);
      }
    }
  }
}

// ---------------------------------------------------------------- flash attention
// Block = 8 waves = one (b,h) and 8 consecutive 16-row q tiles.
// K/V chunks (32 kv) staged in LDS once per block via async copies, double-buffered.
__global__ __launch_bounds__(256)
void attn_wmma(const unsigned short* __restrict__ Qc,
               const unsigned short* __restrict__ Kc,
               const unsigned short* __restrict__ Vt,   // [B,H,D_K,L]
               unsigned short* __restrict__ Oc) {
  __shared__ __align__(16) unsigned short kbuf[2][32 * 64];   // [kv][d]
  __shared__ __align__(16) unsigned short vbuf[2][64 * 32];   // [d][kv]
  __shared__ __align__(16) unsigned short ptile[8][16 * 32];

  const int tid  = threadIdx.x;
  const int lane = tid & 31;
  const int wave = tid >> 5;
  const int col  = lane & 15;
  const int hi   = lane >> 4;

  const int blk    = blockIdx.x;       // 512 = B * H * (L/128)
  const int qgroup = blk & 15;
  const int h      = (blk >> 4) & 15;
  const int b      = blk >> 8;
  const int qtile  = qgroup * 8 + wave;
  const int qbase  = qtile * 16;
  unsigned short* myP = ptile[wave];

  auto issue = [&](int j, int s) {
    {   // K chunk: 32 rows x 128B, 256 lanes x 16B
      const int row = tid >> 3, seg = tid & 7;
      const unsigned short* g =
          Kc + (size_t)(b * Lseq + j * 32 + row) * D_MODEL + h * D_K + seg * 8;
      async_b128(lds_off(&kbuf[s][row * 64 + seg * 8]), g);
    }
    {   // V chunk (already kv-contiguous): 64 rows x 64B, 256 lanes x 16B
      const int row = tid >> 2, seg = tid & 3;
      const unsigned short* g =
          Vt + ((size_t)(b * NUM_HEADS + h) * D_K + row) * Lseq + j * 32 + seg * 8;
      async_b128(lds_off(&vbuf[s][row * 32 + seg * 8]), g);
    }
  };

  // Q A-fragments for d = 0..31 and 32..63
  FragU aQ[2];
  {
    const unsigned short* qrow =
        Qc + (size_t)(b * Lseq + qbase + col) * D_MODEL + h * D_K;
#pragma unroll
    for (int t = 0; t < 2; ++t) {
      const uint4* p = (const uint4*)(qrow + t * 32 + hi * 16);
      aQ[t].q[0] = p[0];
      aQ[t].q[1] = p[1];
    }
  }

  v8f oacc[4] = {{}, {}, {}, {}};
  float mrow[8], lrow[8];
#pragma unroll
  for (int r = 0; r < 8; ++r) { mrow[r] = -1e30f; lrow[r] = 0.0f; }

  const int jmaxBlk = qgroup * 4 + 3;          // covers q rows of whole block
  issue(0, 0);

  for (int j = 0; j <= jmaxBlk; ++j) {
    const int s = j & 1;
    const int kvbase = j * 32;
    wait_async0();           // own async copies for buffer s complete
    __syncthreads();         // all waves' copies visible; prev compute done
    if (j < jmaxBlk) issue(j + 1, s ^ 1);

    if (kvbase <= qbase + 15) {   // wave-uniform causal skip
      // ---- scores S = Q K^T (16 x 32) from LDS
      v8f s0 = {}, s1 = {};
#pragma unroll
      for (int t = 0; t < 2; ++t) {
        FragU bk0, bk1;
        const uint4* p0 = (const uint4*)(&kbuf[s][(col)      * 64 + t * 32 + hi * 16]);
        const uint4* p1 = (const uint4*)(&kbuf[s][(16 + col) * 64 + t * 32 + hi * 16]);
        bk0.q[0] = p0[0]; bk0.q[1] = p0[1];
        bk1.q[0] = p1[0]; bk1.q[1] = p1[1];
        s0 = __builtin_amdgcn_wmma_f32_16x16x32_bf16(false, aQ[t].v, false, bk0.v,
                                                     (short)0, s0, false, false);
        s1 = __builtin_amdgcn_wmma_f32_16x16x32_bf16(false, aQ[t].v, false, bk1.v,
                                                     (short)0, s1, false, false);
      }

      // ---- online softmax
#pragma unroll
      for (int r = 0; r < 8; ++r) {
        const int q = qbase + r + hi * 8;
        float v0 = s0[r] * 0.125f;                 // 1/sqrt(64)
        float v1 = s1[r] * 0.125f;
        if (kvbase + col > q)      v0 = -1e30f;
        if (kvbase + 16 + col > q) v1 = -1e30f;
        float rmax = fmaxf(v0, v1);
#pragma unroll
        for (int off = 1; off < 16; off <<= 1)
          rmax = fmaxf(rmax, __shfl_xor(rmax, off));
        float newm  = fmaxf(mrow[r], rmax);
        float alpha = __expf(mrow[r] - newm);
        float p0 = __expf(v0 - newm);
        float p1 = __expf(v1 - newm);
        float rs = p0 + p1;
#pragma unroll
        for (int off = 1; off < 16; off <<= 1)
          rs += __shfl_xor(rs, off);
        lrow[r] = lrow[r] * alpha + rs;
        mrow[r] = newm;
        oacc[0][r] *= alpha; oacc[1][r] *= alpha;
        oacc[2][r] *= alpha; oacc[3][r] *= alpha;
        myP[(r + hi * 8) * 32 + col]      = f2bf(p0);
        myP[(r + hi * 8) * 32 + 16 + col] = f2bf(p1);
      }

      // ---- reload probs in A-fragment layout
      FragU aP;
      {
        const uint4* p = (const uint4*)(myP + col * 32 + hi * 16);
        aP.q[0] = p[0];
        aP.q[1] = p[1];
      }

      // ---- O += P V  (V tile is [d][kv] in LDS -> contiguous B-frags)
#pragma unroll
      for (int t = 0; t < 4; ++t) {
        FragU bv;
        const uint4* pv = (const uint4*)(&vbuf[s][(t * 16 + col) * 32 + hi * 16]);
        bv.q[0] = pv[0];
        bv.q[1] = pv[1];
        oacc[t] = __builtin_amdgcn_wmma_f32_16x16x32_bf16(false, aP.v, false, bv.v,
                                                          (short)0, oacc[t], false, false);
      }
    }
  }

  // ---- normalize and store (head-concat layout [B*L, 1024])
#pragma unroll
  for (int t = 0; t < 4; ++t) {
#pragma unroll
    for (int r = 0; r < 8; ++r) {
      float val = oacc[t][r] / lrow[r];
      Oc[(size_t)(b * Lseq + qbase + r + hi * 8) * D_MODEL
         + h * D_K + t * 16 + col] = f2bf(val);
    }
  }
}

// ---------------------------------------------------------------- launch
extern "C" void kernel_launch(void* const* d_in, const int* in_sizes, int n_in,
                              void* d_out, int out_size, void* d_ws, size_t ws_size,
                              hipStream_t stream) {
  (void)in_sizes; (void)n_in; (void)out_size; (void)ws_size;
  const float* x  = (const float*)d_in[0];
  const int*  tok = (const int*)d_in[1];
  const float* WQ = (const float*)d_in[2];
  const float* WK = (const float*)d_in[3];
  const float* WV = (const float*)d_in[4];
  const float* WO = (const float*)d_in[5];

  char* ws = (char*)d_ws;
  const size_t MB = 1024u * 1024u;
  unsigned short* xb  = (unsigned short*)(ws);             //  8 MiB
  unsigned short* wqb = (unsigned short*)(ws +  8 * MB);   //  2 MiB each
  unsigned short* wkb = (unsigned short*)(ws + 10 * MB);
  unsigned short* wvb = (unsigned short*)(ws + 12 * MB);
  unsigned short* wob = (unsigned short*)(ws + 14 * MB);
  unsigned short* Qc  = (unsigned short*)(ws + 16 * MB);   //  8 MiB each
  unsigned short* Kc  = (unsigned short*)(ws + 24 * MB);
  unsigned short* Vt  = (unsigned short*)(ws + 32 * MB);   // [B,H,D_K,L]
  unsigned short* Oc  = (unsigned short*)(ws + 40 * MB);   // ends at 48 MiB

  const int nX = Mrows * D_MODEL;
  const int nW = D_MODEL * D_MODEL;
  cast_f32_bf16<<<(nX + 255) / 256, 256, 0, stream>>>(x,  xb,  nX);
  cast_f32_bf16<<<(nW + 255) / 256, 256, 0, stream>>>(WQ, wqb, nW);
  cast_f32_bf16<<<(nW + 255) / 256, 256, 0, stream>>>(WK, wkb, nW);
  cast_f32_bf16<<<(nW + 255) / 256, 256, 0, stream>>>(WV, wvb, nW);
  cast_f32_bf16<<<(nW + 255) / 256, 256, 0, stream>>>(WO, wob, nW);

  dim3 ggrid(Mrows / 16, D_MODEL / (64 * 8));              // (256, 2), 8 waves/block
  gemm_wmma<<<ggrid, 256, 0, stream>>>(xb, wqb, Qc, tok, 1);   // Q + RoPE
  gemm_wmma<<<ggrid, 256, 0, stream>>>(xb, wkb, Kc, tok, 1);   // K + RoPE
  gemm_wmma<<<ggrid, 256, 0, stream>>>(xb, wvb, Vt, tok, 3);   // V (transposed)

  attn_wmma<<<Bsz * NUM_HEADS * (Lseq / 128), 256, 0, stream>>>(Qc, Kc, Vt, Oc);

  gemm_wmma<<<ggrid, 256, 0, stream>>>(Oc, wob, d_out, tok, 2); // out proj (f32)
}
